// AGNNConv_6932077216252
// MI455X (gfx1250) — matrix operand order repaired and checked by reading, not compiled, not run on hardware
//
#include <hip/hip_runtime.h>
#include <hip/hip_fp16.h>

typedef __attribute__((ext_vector_type(16))) _Float16 v16h;
typedef __attribute__((ext_vector_type(8)))  float    v8f;

#define N_NODES 50000
#define N_EDGES 800000
#define D_FEAT  128

// ---- order-preserving float<->int mapping for atomicMax on floats ----
__device__ __forceinline__ int f2ord(float f) {
    int b = __float_as_int(f);
    return b >= 0 ? b : ((~b) ^ 0x80000000);
}
__device__ __forceinline__ float ord2f(int s) {
    int b = (s >= 0) ? s : ~(s ^ 0x80000000);
    return __int_as_float(b);
}

// ---- kernel 1: inverse L2 norm, one wave (32 lanes) per node ----
__global__ void __launch_bounds__(256) k_invnorm(const float* __restrict__ feat,
                                                 float* __restrict__ inv) {
    int lane = threadIdx.x & 31;
    int node = (blockIdx.x * blockDim.x + threadIdx.x) >> 5;
    if (node >= N_NODES) return;
    const float4* row = (const float4*)(feat + (size_t)node * D_FEAT);
    float4 v = row[lane];                       // 4 floats per lane, 128 total
    float ss = v.x * v.x + v.y * v.y + v.z * v.z + v.w * v.w;
    #pragma unroll
    for (int o = 16; o > 0; o >>= 1) ss += __shfl_xor(ss, o, 32);
    float nrm = fmaxf(sqrtf(ss), 1e-12f);
    if (lane == 0) inv[node] = 1.0f / nrm;
}

// ---- kernel 2: zero out[], init max/denom accumulators ----
__global__ void __launch_bounds__(256) k_init(float* __restrict__ out,
                                              int* __restrict__ mord,
                                              float* __restrict__ denom) {
    long long i = (long long)blockIdx.x * blockDim.x + threadIdx.x;
    if (i < (long long)N_NODES * D_FEAT) out[i] = 0.0f;
    if (i < N_NODES) {
        mord[i]  = f2ord(-__builtin_inff());
        denom[i] = 0.0f;
    }
}

__device__ __forceinline__ void pack8(v16h& v, int base, float4 p, float4 q) {
    v[base + 0] = (_Float16)p.x; v[base + 1] = (_Float16)p.y;
    v[base + 2] = (_Float16)p.z; v[base + 3] = (_Float16)p.w;
    v[base + 4] = (_Float16)q.x; v[base + 5] = (_Float16)q.y;
    v[base + 6] = (_Float16)q.z; v[base + 7] = (_Float16)q.w;
}

// ---- kernel 3: edge logits via WMMA, one wave handles 16 edges ----
// A tile: 16 gathered src rows (16x128), B tile: 16 gathered dst rows as columns
// (128x16). Four v_wmma_f32_16x16x32_f16 accumulate C; diag(C) = 16 edge dots.
__global__ void __launch_bounds__(256) k_edge_logits(
        const float* __restrict__ feat, const float* __restrict__ inv,
        const int* __restrict__ src, const int* __restrict__ dst,
        const float* __restrict__ beta,
        float* __restrict__ elog, int* __restrict__ mord) {
    int lane = threadIdx.x & 31;
    int wave = (blockIdx.x * blockDim.x + threadIdx.x) >> 5;
    int eb = wave * 16;
    if (eb >= N_EDGES) return;              // wave-uniform: EXEC stays all-ones

    int m  = lane & 15;                     // this lane's edge within the tile
    int hi = lane >> 4;                     // lane half selects K sub-blocks
    int s  = src[eb + m];
    int d  = dst[eb + m];
    const float* arow = feat + (size_t)s * D_FEAT;   // A row M=m
    const float* brow = feat + (size_t)d * D_FEAT;   // B column N=m

    v8f c = {};
    #pragma unroll
    for (int t = 0; t < 4; ++t) {
        int kb = t * 32;
        // A 16x32 f16 layout: lanes 0-15 hold K=kb+[0..7] (V0-3), kb+[16..23] (V4-7);
        // lanes 16-31 hold K=kb+[8..15], kb+[24..31].
        const float4* a0 = (const float4*)(arow + kb + hi * 8);
        const float4* a1 = (const float4*)(arow + kb + 16 + hi * 8);
        float4 af0 = a0[0], af1 = a0[1], af2 = a1[0], af3 = a1[1];
        // B 32x16 f16 layout: lanes 0-15 (col N) hold K=kb+[0..7], kb+[16..23]; lanes 16-31 K=kb+[8..15], kb+[24..31].
        const float4* b0 = (const float4*)(brow + kb + hi * 8);
        const float4* b1 = (const float4*)(brow + kb + 16 + hi * 8);
        float4 bf0 = b0[0], bf1 = b0[1], bf2 = b1[0], bf3 = b1[1];

        v16h a, b;
        pack8(a, 0, af0, af1); pack8(a, 8, af2, af3);
        pack8(b, 0, bf0, bf1); pack8(b, 8, bf2, bf3);
        c = __builtin_amdgcn_wmma_f32_16x16x32_f16(
                /*neg_a=*/false, a, /*neg_b=*/false, b,
                /*c_mod=*/(short)0, c, /*reuse_a=*/false, /*reuse_b=*/false);
    }

    // Diagonal extraction: C/D layout -> diag m<8 lives in lane m, VGPR m;
    // diag m>=8 lives in lane m+16, VGPR m-8. Each writer lane owns edge m.
    bool wlo = lane < 8, whi = lane >= 24;
    if (wlo || whi) {
        int r = wlo ? lane : (lane - 24);
        float dotv = 0.0f;
        #pragma unroll
        for (int i = 0; i < 8; ++i) if (r == i) dotv = c[i];
        float ev = beta[0] * dotv * inv[s] * inv[d];
        elog[eb + m] = ev;
        atomicMax(&mord[d], f2ord(ev));
    }
}

// ---- kernel 4: e_exp = exp(e - m[dst]); denom[dst] += e_exp ----
__global__ void __launch_bounds__(256) k_edge_exp(const int* __restrict__ dst,
                                                  const int* __restrict__ mord,
                                                  float* __restrict__ elog,
                                                  float* __restrict__ denom) {
    int i = blockIdx.x * blockDim.x + threadIdx.x;
    if (i >= N_EDGES) return;
    int d = dst[i];
    float ex = expf(elog[i] - ord2f(mord[d]));
    elog[i] = ex;                 // in-place: edge i is the sole reader/writer
    atomicAdd(&denom[d], ex);
}

// ---- kernel 5: out[dst] += feat[src] * p, one wave per edge ----
__global__ void __launch_bounds__(256) k_aggregate(
        const float* __restrict__ feat, const int* __restrict__ src,
        const int* __restrict__ dst, const float* __restrict__ elog,
        const float* __restrict__ denom, float* __restrict__ out) {
    int lane = threadIdx.x & 31;
    int e = (blockIdx.x * blockDim.x + threadIdx.x) >> 5;
    if (e >= N_EDGES) return;
    int s = src[e], d = dst[e];
    float p = elog[e] / denom[d];
    const float4* frow = (const float4*)(feat + (size_t)s * D_FEAT);
    float4 v = frow[lane];
    float* orow = out + (size_t)d * D_FEAT + lane * 4;
    atomicAdd(orow + 0, v.x * p);
    atomicAdd(orow + 1, v.y * p);
    atomicAdd(orow + 2, v.z * p);
    atomicAdd(orow + 3, v.w * p);
}

extern "C" void kernel_launch(void* const* d_in, const int* in_sizes, int n_in,
                              void* d_out, int out_size, void* d_ws, size_t ws_size,
                              hipStream_t stream) {
    const float* feat = (const float*)d_in[0];   // [N, 128]
    const float* beta = (const float*)d_in[1];   // [1]
    const int*   src  = (const int*)d_in[2];     // [E]
    const int*   dst  = (const int*)d_in[3];     // [E]
    float*       out  = (float*)d_out;           // [N, 128]

    char* ws = (char*)d_ws;
    float* elog  = (float*)ws;                                   // E floats
    float* inv   = (float*)(ws + (size_t)N_EDGES * 4);           // N floats
    int*   mord  = (int*)  (ws + (size_t)N_EDGES * 4 + (size_t)N_NODES * 4);
    float* denom = (float*)(ws + (size_t)N_EDGES * 4 + (size_t)N_NODES * 8);

    // 1) per-node inverse norms: 50000 waves, 8 waves/block
    k_invnorm<<<(N_NODES + 7) / 8, 256, 0, stream>>>(feat, inv);
    // 2) zero out[] and init accumulators (poison-safe per call)
    k_init<<<(N_NODES * D_FEAT + 255) / 256, 256, 0, stream>>>(out, mord, denom);
    // 3) WMMA edge logits: 16 edges/wave -> 50000 waves
    k_edge_logits<<<(N_EDGES / 16 + 7) / 8, 256, 0, stream>>>(
        feat, inv, src, dst, beta, elog, mord);
    // 4) exp + segment-sum denominator
    k_edge_exp<<<(N_EDGES + 255) / 256, 256, 0, stream>>>(dst, mord, elog, denom);
    // 5) weighted scatter-add aggregation: one wave per edge
    k_aggregate<<<(N_EDGES * 32 + 255) / 256, 256, 0, stream>>>(
        feat, src, dst, elog, denom, out);
}
